// MultiHeadSelfAttention_27513560499019
// MI455X (gfx1250) — matrix operand order repaired and checked by reading, not compiled, not run on hardware
//
#include <hip/hip_runtime.h>

// ---------------------------------------------------------------------------
// MI455X (gfx1250) multi-head self-attention, f16 WMMA path, f32 accumulate.
// B=2, S=2048, D=1024, H=16, HD=64
// LDS-staged operand reuse, 64-deep slabs, register double-buffering,
// 32 rows per wave (2 A-tiles) so every LDS B-tile feeds two WMMAs.
// ---------------------------------------------------------------------------

#define B_  2
#define S_  2048
#define D_  1024
#define H_  16
#define HD_ 64

// padded LDS row strides (halves): 144B / 80B strides spread the 16 lanes of a
// ds_load_b128 across distinct bank groups
#define BSTR 72   // weight slab rows (64 d + pad)
#define KSTR 72   // K slab rows (64 e + pad)
#define VSTR 72   // V slab rows (64 t + pad)
#define PSTR 40   // P tile rows (32 t + pad)

typedef __attribute__((ext_vector_type(16))) _Float16 v16h;
typedef __attribute__((ext_vector_type(8)))  _Float16 v8h;
typedef __attribute__((ext_vector_type(8)))  float    v8f;

union V16 { v16h v; v8h h[2]; };

__device__ inline v8f wmma_f16(v16h a, v16h b, v8f c) {
    // D = A(16x32 f16) * B(32x16 f16) + C(16x16 f32)
    return __builtin_amdgcn_wmma_f32_16x16x32_f16(
        /*neg_a=*/false, a, /*neg_b=*/false, b,
        /*c_mod=*/(short)0, c, /*reuse_a=*/false, /*reuse_b=*/false);
}

// ---------------------------------------------------------------------------
// Kernel 0a: x (f32) -> xh (f16), same layout [B,S,D]
// ---------------------------------------------------------------------------
__global__ void cvt_x_kernel(const float* __restrict__ x,
                             _Float16* __restrict__ xh, int n) {
    int i = blockIdx.x * blockDim.x + threadIdx.x;
    if (i < n) xh[i] = (_Float16)x[i];
}

// ---------------------------------------------------------------------------
// Kernel 0b: W_{q,k,v} [H,D,HD] f32 -> wt [3,H,HD,D] f16 (transposed so the
// WMMA B-matrix lane chunks are contiguous along the GEMM K dimension d).
// ---------------------------------------------------------------------------
__global__ void cvt_w_kernel(const float* __restrict__ wq,
                             const float* __restrict__ wk,
                             const float* __restrict__ wv,
                             _Float16* __restrict__ wt, int n) {
    int o = blockIdx.x * blockDim.x + threadIdx.x;
    if (o >= n) return;
    int d  = o & (D_ - 1);
    int r  = o >> 10;            // D_ = 1024
    int e  = r & (HD_ - 1);
    int r2 = r >> 6;             // HD_ = 64
    int h  = r2 & (H_ - 1);
    int m  = r2 >> 4;            // H_ = 16
    const float* w = (m == 0) ? wq : (m == 1) ? wk : wv;
    wt[o] = (_Float16)w[((size_t)h * D_ + d) * HD_ + e];
}

// ---------------------------------------------------------------------------
// Kernel 1: QKV projection. Block = 4 waves = one 128(s) x 64(e) tile of one
// matrix for one (b,h); each wave owns 32 s-rows (two A-tiles). Weight slab
// (64e x 64d, 8KB) staged through LDS per k-stage, shared by all 4 waves and
// reused by both A-tiles: 16 WMMAs per wave per barrier pair.
//   grid = (B*H*(S/128), 3) = (512, 3), block = 128
//   Q -> [B,H,S,HD] f16, pre-scaled by 1/sqrt(HD)
//   K -> [B,H,S,HD] f16
//   V -> [B,H,HD,S] f16 (transposed)
// ---------------------------------------------------------------------------
__global__ __launch_bounds__(128)
void qkv_proj_kernel(const _Float16* __restrict__ xh,
                     const _Float16* __restrict__ wt,
                     _Float16* __restrict__ Q,
                     _Float16* __restrict__ K,
                     _Float16* __restrict__ Vt) {
    __shared__ __align__(16) _Float16 ldsB[64 * BSTR];

    const int tid  = threadIdx.x;
    const int wave = tid >> 5;
    const int lane = tid & 31;
    const int hi   = lane >> 4;
    const int lcol = lane & 15;

    const int mat = blockIdx.y;              // 0=Q 1=K 2=V
    const int sb  = blockIdx.x & 15;         // S/128 = 16
    const int h   = (blockIdx.x >> 4) & 15;
    const int b   = blockIdx.x >> 8;
    const int sw  = sb * 128 + wave * 32;    // this wave: rows sw..sw+31

    // A rows (two 16-row tiles), contiguous in d
    const _Float16* arow0 = xh + (size_t)(b * S_ + sw + lcol) * D_;
    const _Float16* arow1 = arow0 + (size_t)16 * D_;
    // weight base for this (mat, h): rows e, contiguous in d
    const _Float16* wbase = wt + (size_t)((mat * H_ + h) * HD_) * D_;

    // staging: slab = 64 e-rows x 64 d = 512 chunks of 8 halves (16B);
    // chunk c: e = c>>3, q = c&7; thread owns rows se, se+16, se+32, se+48.
    const int se = tid >> 3, sq = tid & 7;

    v8f acc[2][4];
    #pragma unroll
    for (int r = 0; r < 2; ++r)
        #pragma unroll
        for (int j = 0; j < 4; ++j) acc[r][j] = (v8f){};

    // prologue: prefetch slab 0
    v8h r0 = *(const v8h*)(wbase + (size_t)(se)      * D_ + sq * 8);
    v8h r1 = *(const v8h*)(wbase + (size_t)(se + 16) * D_ + sq * 8);
    v8h r2 = *(const v8h*)(wbase + (size_t)(se + 32) * D_ + sq * 8);
    v8h r3 = *(const v8h*)(wbase + (size_t)(se + 48) * D_ + sq * 8);

    #pragma clang loop unroll(disable)
    for (int kk = 0; kk < D_ / 64; ++kk) {   // 16 stages
        const int d0 = kk * 64;
        __syncthreads();                       // prior reads of ldsB done
        *(v8h*)(&ldsB[(se)      * BSTR + sq * 8]) = r0;
        *(v8h*)(&ldsB[(se + 16) * BSTR + sq * 8]) = r1;
        *(v8h*)(&ldsB[(se + 32) * BSTR + sq * 8]) = r2;
        *(v8h*)(&ldsB[(se + 48) * BSTR + sq * 8]) = r3;
        __syncthreads();                       // slab visible

        if (kk + 1 < D_ / 64) {                // prefetch next slab (overlaps)
            r0 = *(const v8h*)(wbase + (size_t)(se)      * D_ + d0 + 64 + sq * 8);
            r1 = *(const v8h*)(wbase + (size_t)(se + 16) * D_ + d0 + 64 + sq * 8);
            r2 = *(const v8h*)(wbase + (size_t)(se + 32) * D_ + d0 + 64 + sq * 8);
            r3 = *(const v8h*)(wbase + (size_t)(se + 48) * D_ + d0 + 64 + sq * 8);
        }

        #pragma unroll
        for (int dd = 0; dd < 64; dd += 32) {
            V16 a0, a1;
            a0.h[0] = *(const v8h*)(arow0 + d0 + dd + hi * 8);
            a0.h[1] = *(const v8h*)(arow0 + d0 + dd + 16 + hi * 8);
            a1.h[0] = *(const v8h*)(arow1 + d0 + dd + hi * 8);
            a1.h[1] = *(const v8h*)(arow1 + d0 + dd + 16 + hi * 8);
            V16 bm[4];                         // all tiles first: partial waits
            #pragma unroll
            for (int et = 0; et < 4; ++et) {
                const _Float16* bp = &ldsB[(et * 16 + lcol) * BSTR + dd];
                bm[et].h[0] = *(const v8h*)(bp + hi * 8);
                bm[et].h[1] = *(const v8h*)(bp + 16 + hi * 8);
            }
            #pragma unroll
            for (int et = 0; et < 4; ++et) {   // each B tile feeds 2 WMMAs
                acc[0][et] = wmma_f16(a0.v, bm[et].v, acc[0][et]);
                acc[1][et] = wmma_f16(a1.v, bm[et].v, acc[1][et]);
            }
        }
    }

    // epilogue: this wave owns rows sw..sw+31, all 64 e
    #pragma unroll
    for (int r = 0; r < 2; ++r) {
        const int swr = sw + r * 16;
        if (mat == 0) {
            _Float16* q = Q + (size_t)((b * H_ + h) * S_ + swr) * HD_;
            #pragma unroll
            for (int et = 0; et < 4; ++et)
                #pragma unroll
                for (int i = 0; i < 8; ++i)
                    q[(size_t)(i + hi * 8) * HD_ + et * 16 + lcol] =
                        (_Float16)(acc[r][et][i] * 0.125f);   // 1/sqrt(64)
        } else if (mat == 1) {
            _Float16* kp = K + (size_t)((b * H_ + h) * S_ + swr) * HD_;
            #pragma unroll
            for (int et = 0; et < 4; ++et)
                #pragma unroll
                for (int i = 0; i < 8; ++i)
                    kp[(size_t)(i + hi * 8) * HD_ + et * 16 + lcol] =
                        (_Float16)acc[r][et][i];
        } else {
            _Float16* vp = Vt + (size_t)((b * H_ + h) * HD_) * S_ + swr;
            #pragma unroll
            for (int et = 0; et < 4; ++et)
                #pragma unroll
                for (int i = 0; i < 8; ++i)
                    vp[(size_t)(et * 16 + lcol) * S_ + i + hi * 8] =
                        (_Float16)acc[r][et][i];
        }
    }
}

// ---------------------------------------------------------------------------
// Kernel 2: flash attention. Block = 4 waves = 128 query rows of one (b,h);
// each wave owns 32 q-rows (two A-tiles). K (64t x 64e) and V (64e x 64t)
// slabs staged through LDS per 64-key stage, shared by all 4 waves and reused
// by both A-tiles: 32 WMMAs per wave per barrier pair.
//   grid = B*H*(S/128) = 512, block = 128
// ---------------------------------------------------------------------------
__global__ __launch_bounds__(128)
void attn_kernel(const _Float16* __restrict__ Q,
                 const _Float16* __restrict__ K,
                 const _Float16* __restrict__ Vt,
                 float* __restrict__ out) {
    __shared__ __align__(16) _Float16 ldsK[64 * KSTR];
    __shared__ __align__(16) _Float16 ldsV[64 * VSTR];
    __shared__ __align__(16) _Float16 ldsP[4][32 * PSTR];  // per-wave, 2 tiles

    const int tid  = threadIdx.x;
    const int wave = tid >> 5;
    const int lane = tid & 31;
    const int hi   = lane >> 4;
    const int lcol = lane & 15;

    const int qb = blockIdx.x & 15;          // S/128 = 16
    const int h  = (blockIdx.x >> 4) & 15;
    const int b  = blockIdx.x >> 8;
    const int sbase = qb * 128 + wave * 32;  // this wave: rows sbase..sbase+31

    // Q A-tiles: 2 row-tiles x 2 K-halves, loaded once.
    V16 aq[2][2];
    #pragma unroll
    for (int r = 0; r < 2; ++r) {
        const _Float16* qrow =
            Q + (size_t)((b * H_ + h) * S_ + sbase + r * 16 + lcol) * HD_;
        aq[r][0].h[0] = *(const v8h*)(qrow + hi * 8);
        aq[r][0].h[1] = *(const v8h*)(qrow + 16 + hi * 8);
        aq[r][1].h[0] = *(const v8h*)(qrow + 32 + hi * 8);
        aq[r][1].h[1] = *(const v8h*)(qrow + 48 + hi * 8);
    }

    const _Float16* Kb = K  + (size_t)(b * H_ + h) * S_ * HD_;
    const _Float16* Vb = Vt + (size_t)(b * H_ + h) * HD_ * S_;

    // staging: both slabs are 64 rows x 64 halves = 512 chunks of 8 halves;
    // chunk c: row = c>>3, q = c&7; thread owns rows sr, sr+16, sr+32, sr+48.
    const int sr = tid >> 3, sq = tid & 7;

    float mrun[2][8], lrun[2][8];
    v8f acc[2][4];
    #pragma unroll
    for (int r = 0; r < 2; ++r) {
        #pragma unroll
        for (int i = 0; i < 8; ++i) { mrun[r][i] = -1e30f; lrun[r][i] = 0.0f; }
        #pragma unroll
        for (int j = 0; j < 4; ++j) acc[r][j] = (v8f){};
    }

    // prologue: prefetch slab for key-stage 0
    v8h kr0 = *(const v8h*)(Kb + (size_t)(sr)      * HD_ + sq * 8);
    v8h kr1 = *(const v8h*)(Kb + (size_t)(sr + 16) * HD_ + sq * 8);
    v8h kr2 = *(const v8h*)(Kb + (size_t)(sr + 32) * HD_ + sq * 8);
    v8h kr3 = *(const v8h*)(Kb + (size_t)(sr + 48) * HD_ + sq * 8);
    v8h vr0 = *(const v8h*)(Vb + (size_t)(sr)      * S_ + sq * 8);
    v8h vr1 = *(const v8h*)(Vb + (size_t)(sr + 16) * S_ + sq * 8);
    v8h vr2 = *(const v8h*)(Vb + (size_t)(sr + 32) * S_ + sq * 8);
    v8h vr3 = *(const v8h*)(Vb + (size_t)(sr + 48) * S_ + sq * 8);

    #pragma clang loop unroll(disable)
    for (int kb = 0; kb < S_ / 64; ++kb) {   // 32 stages of 64 keys
        const int t0 = kb * 64;
        __syncthreads();                       // prior reads of ldsK/ldsV done
        *(v8h*)(&ldsK[(sr)      * KSTR + sq * 8]) = kr0;
        *(v8h*)(&ldsK[(sr + 16) * KSTR + sq * 8]) = kr1;
        *(v8h*)(&ldsK[(sr + 32) * KSTR + sq * 8]) = kr2;
        *(v8h*)(&ldsK[(sr + 48) * KSTR + sq * 8]) = kr3;
        *(v8h*)(&ldsV[(sr)      * VSTR + sq * 8]) = vr0;
        *(v8h*)(&ldsV[(sr + 16) * VSTR + sq * 8]) = vr1;
        *(v8h*)(&ldsV[(sr + 32) * VSTR + sq * 8]) = vr2;
        *(v8h*)(&ldsV[(sr + 48) * VSTR + sq * 8]) = vr3;
        __syncthreads();                       // slab visible

        if (kb + 1 < S_ / 64) {                // prefetch next slab (overlaps)
            kr0 = *(const v8h*)(Kb + (size_t)(t0 + 64 + sr)      * HD_ + sq * 8);
            kr1 = *(const v8h*)(Kb + (size_t)(t0 + 64 + sr + 16) * HD_ + sq * 8);
            kr2 = *(const v8h*)(Kb + (size_t)(t0 + 64 + sr + 32) * HD_ + sq * 8);
            kr3 = *(const v8h*)(Kb + (size_t)(t0 + 64 + sr + 48) * HD_ + sq * 8);
            vr0 = *(const v8h*)(Vb + (size_t)(sr)      * S_ + t0 + 64 + sq * 8);
            vr1 = *(const v8h*)(Vb + (size_t)(sr + 16) * S_ + t0 + 64 + sq * 8);
            vr2 = *(const v8h*)(Vb + (size_t)(sr + 32) * S_ + t0 + 64 + sq * 8);
            vr3 = *(const v8h*)(Vb + (size_t)(sr + 48) * S_ + t0 + 64 + sq * 8);
        }

        #pragma unroll
        for (int sb = 0; sb < 64; sb += 32) {  // two 32-key sub-blocks
            // ---- scores: 32 x 32 block of Q K^T (scale folded into Q) ----
            V16 bk[4];                         // all tiles first: partial waits
            const _Float16* k0 = &ldsK[(sb + lcol) * KSTR];
            const _Float16* k1 = &ldsK[(sb + 16 + lcol) * KSTR];
            bk[0].h[0] = *(const v8h*)(k0 + hi * 8);
            bk[0].h[1] = *(const v8h*)(k0 + 16 + hi * 8);
            bk[1].h[0] = *(const v8h*)(k0 + 32 + hi * 8);
            bk[1].h[1] = *(const v8h*)(k0 + 48 + hi * 8);
            bk[2].h[0] = *(const v8h*)(k1 + hi * 8);
            bk[2].h[1] = *(const v8h*)(k1 + 16 + hi * 8);
            bk[3].h[0] = *(const v8h*)(k1 + 32 + hi * 8);
            bk[3].h[1] = *(const v8h*)(k1 + 48 + hi * 8);
            v8f s[2][2];
            #pragma unroll
            for (int r = 0; r < 2; ++r) {      // each B tile feeds 2 WMMAs
                s[r][0] = (v8f){};
                s[r][1] = (v8f){};
                s[r][0] = wmma_f16(aq[r][0].v, bk[0].v, s[r][0]);
                s[r][0] = wmma_f16(aq[r][1].v, bk[1].v, s[r][0]);
                s[r][1] = wmma_f16(aq[r][0].v, bk[2].v, s[r][1]);
                s[r][1] = wmma_f16(aq[r][1].v, bk[3].v, s[r][1]);
            }

            // ---- online softmax per row-tile ----
            #pragma unroll
            for (int r = 0; r < 2; ++r) {
                float corr[8];
                #pragma unroll
                for (int i = 0; i < 8; ++i) {
                    float rv = fmaxf(s[r][0][i], s[r][1][i]);
                    rv = fmaxf(rv, __shfl_xor(rv, 1, 32));
                    rv = fmaxf(rv, __shfl_xor(rv, 2, 32));
                    rv = fmaxf(rv, __shfl_xor(rv, 4, 32));
                    rv = fmaxf(rv, __shfl_xor(rv, 8, 32));
                    const float mnew = fmaxf(mrun[r][i], rv);
                    corr[i] = __expf(mrun[r][i] - mnew);
                    mrun[r][i] = mnew;
                    const float p0 = __expf(s[r][0][i] - mnew);
                    const float p1 = __expf(s[r][1][i] - mnew);
                    s[r][0][i] = p0; s[r][1][i] = p1;
                    float rs = p0 + p1;
                    rs += __shfl_xor(rs, 1, 32);
                    rs += __shfl_xor(rs, 2, 32);
                    rs += __shfl_xor(rs, 4, 32);
                    rs += __shfl_xor(rs, 8, 32);
                    lrun[r][i] = lrun[r][i] * corr[i] + rs;
                }
                #pragma unroll
                for (int j = 0; j < 4; ++j)
                    #pragma unroll
                    for (int i = 0; i < 8; ++i)
                        acc[r][j][i] *= corr[i];

                // lane transpose of P tile r (C-layout -> A-layout)
                _Float16* pw = &ldsP[wave][r * 16 * PSTR];
                #pragma unroll
                for (int i = 0; i < 8; ++i) {
                    const int rr = i + hi * 8;
                    pw[rr * PSTR + lcol]      = (_Float16)s[r][0][i];
                    pw[rr * PSTR + 16 + lcol] = (_Float16)s[r][1][i];
                }
            }
            V16 pa[2];   // same-wave LDS is in-order; no barrier needed
            #pragma unroll
            for (int r = 0; r < 2; ++r) {
                const _Float16* pr = &ldsP[wave][r * 16 * PSTR + lcol * PSTR];
                pa[r].h[0] = *(const v8h*)(pr + hi * 8);
                pa[r].h[1] = *(const v8h*)(pr + 16 + hi * 8);
            }

            // ---- acc += P (32x32) * V (32x64) from LDS ----
            V16 bv[4];
            #pragma unroll
            for (int j = 0; j < 4; ++j) {
                const _Float16* vp = &ldsV[(j * 16 + lcol) * VSTR + sb];
                bv[j].h[0] = *(const v8h*)(vp + hi * 8);
                bv[j].h[1] = *(const v8h*)(vp + 16 + hi * 8);
            }
            #pragma unroll
            for (int j = 0; j < 4; ++j) {      // each B tile feeds 2 WMMAs
                acc[0][j] = wmma_f16(pa[0].v, bv[j].v, acc[0][j]);
                acc[1][j] = wmma_f16(pa[1].v, bv[j].v, acc[1][j]);
            }
        }
    }

    // ---- epilogue: normalize and write fp32 output [B,S,H*HD] ----
    #pragma unroll
    for (int r = 0; r < 2; ++r) {
        float* orow = out + (size_t)(b * S_ + sbase + r * 16) * D_ + h * HD_;
        #pragma unroll
        for (int i = 0; i < 8; ++i) {
            const float inv = 1.0f / lrun[r][i];
            #pragma unroll
            for (int j = 0; j < 4; ++j)
                orow[(size_t)(i + hi * 8) * D_ + j * 16 + lcol] = acc[r][j][i] * inv;
        }
    }
}

// ---------------------------------------------------------------------------
// Launch: cvt_x -> cvt_w -> qkv_proj -> attn, all stream-ordered.
// Workspace (f16 elems): xh 4.19M | wt 3.15M | Q 4.19M | K 4.19M | Vt 4.19M
//   ~40 MB total.
// ---------------------------------------------------------------------------
extern "C" void kernel_launch(void* const* d_in, const int* in_sizes, int n_in,
                              void* d_out, int out_size, void* d_ws, size_t ws_size,
                              hipStream_t stream) {
    const float* x  = (const float*)d_in[0];
    const float* wq = (const float*)d_in[1];
    const float* wk = (const float*)d_in[2];
    const float* wv = (const float*)d_in[3];
    float* out = (float*)d_out;

    _Float16* xh = (_Float16*)d_ws;
    _Float16* wt = xh + (size_t)B_ * S_ * D_;
    _Float16* Q  = wt + (size_t)3 * H_ * HD_ * D_;
    _Float16* K  = Q  + (size_t)B_ * H_ * S_ * HD_;
    _Float16* Vt = K  + (size_t)B_ * H_ * S_ * HD_;

    const int nX = B_ * S_ * D_;            // 4194304
    const int nW = 3 * H_ * HD_ * D_;       // 3145728

    cvt_x_kernel<<<nX / 256, 256, 0, stream>>>(x, xh, nX);
    cvt_w_kernel<<<nW / 256, 256, 0, stream>>>(wq, wk, wv, wt, nW);
    qkv_proj_kernel<<<dim3(B_ * H_ * (S_ / 128), 3), 128, 0, stream>>>(xh, wt, Q, K, Vt);
    attn_kernel<<<B_ * H_ * (S_ / 128), 128, 0, stream>>>(Q, K, Vt, out);
}
